// HybridFusionModel_26637387170019
// MI455X (gfx1250) — matrix operand order repaired and checked by reading, not compiled, not run on hardware
//
#include <hip/hip_runtime.h>
#include <hip/hip_bf16.h>
#include <stdint.h>

typedef __bf16 bf16_t;
typedef __attribute__((ext_vector_type(16))) bf16_t v16bf;
typedef __attribute__((ext_vector_type(8)))  float  v8f;
typedef __attribute__((ext_vector_type(4)))  unsigned int u32x4;  // raw 128-bit vector

// explicit global-address-space pointers so loads lower to global_load_* (LOADcnt
// only) instead of flat_load_* (LOADcnt+DScnt, which serializes against LDS ops)
typedef const u32x4  __attribute__((address_space(1)))* g_cu4;
typedef const float  __attribute__((address_space(1)))* g_cf32;
typedef const bf16_t __attribute__((address_space(1)))* g_cbf16;
typedef float        __attribute__((address_space(1)))* g_f32;
typedef bf16_t       __attribute__((address_space(1)))* g_bf16;

__device__ __forceinline__ float gloadf(const float* p)  { return *(g_cf32)(uintptr_t)p; }
__device__ __forceinline__ u32x4 gload4(const void* p)   { return *(g_cu4)(uintptr_t)p; }
__device__ __forceinline__ float gloadbf(const bf16_t* p){ return (float)*(g_cbf16)(uintptr_t)p; }
__device__ __forceinline__ void  gstoref(float* p, float v) { *(g_f32)(uintptr_t)p = v; }

#define HID  512
#define MT   32      // batch rows per workgroup
#define S    520     // padded LDS stride (elements) for 512-wide buffers
#define TSTR 776     // padded stride for normalized text (768+8)
#define NSTR 72      // padded stride for normalized num  (64+8)

enum { EP_NONE, EP_RELU, EP_SIG, EP_ADD };

// A-fragment (16x32 bf16) from LDS: lane holds row m0+lane%16;
// low half-wave: K[k0..k0+7] then K[k0+16..k0+23]; high half-wave offset +8.
__device__ __forceinline__ v16bf load_a(const bf16_t* base, int stride, int m, int k0, int lane) {
  const bf16_t* p = base + m * stride + k0 + ((lane & 16) ? 8 : 0);
  union { v16bf v; u32x4 q[2]; } f;
  f.q[0] = *(const u32x4*)(p);
  f.q[1] = *(const u32x4*)(p + 16);
  return f.v;
}

// B-fragment (32x16 bf16, weight row W[n][k]): lane holds column n;
// low half-wave K[k0..k0+7] then K[k0+16..k0+23]; high half-wave offset +8. Global (L2) load.
__device__ __forceinline__ v16bf load_b(const bf16_t* W, int K, int n, int k0, int lane) {
  const bf16_t* p = W + (size_t)n * (size_t)K + k0 + ((lane & 16) ? 8 : 0);
  union { v16bf v; u32x4 q[2]; } f;
  f.q[0] = gload4(p);
  f.q[1] = gload4(p + 16);
  return f.v;
}

// 32xKTOT (A in LDS, optionally split across two buffers at KSPLIT for concat
// inputs) x KTOTx512 (bf16 weight rows in L2) -> 32x512 bf16 in LDS with fused
// bias + epilogue. 8 waves; each wave owns one 64-col group across BOTH 16-row
// strips: 8 accumulators, each B-fragment feeds 2 WMMAs (loaded once).
template<int KTOT, int KSPLIT, int MODE>
__device__ void gemm_tile(const bf16_t* Xa, int sXa, const bf16_t* Xb, int sXb,
                          const bf16_t* __restrict__ W, const float* __restrict__ bias,
                          bf16_t* D, int sD, int tid)
{
  const int lane = tid & 31, wave = tid >> 5;
  const int lm = lane & 15, lh = lane >> 4;
  const int n0 = wave * 64;

  v8f acc[2][4];
#pragma unroll
  for (int s = 0; s < 2; ++s)
#pragma unroll
    for (int j = 0; j < 4; ++j)
#pragma unroll
      for (int r = 0; r < 8; ++r) acc[s][j][r] = 0.0f;

  for (int kk = 0; kk < KTOT; kk += 32) {
    const bf16_t* Ab; int sA, kl;
    if (KSPLIT >= KTOT || kk < KSPLIT) { Ab = Xa; sA = sXa; kl = kk; }
    else                               { Ab = Xb; sA = sXb; kl = kk - KSPLIT; }
    v16bf a0 = load_a(Ab, sA, lm,      kl, lane);
    v16bf a1 = load_a(Ab, sA, 16 + lm, kl, lane);
#pragma unroll
    for (int j = 0; j < 4; ++j) {
      v16bf b = load_b(W, KTOT, n0 + j * 16, kk, lane);
      acc[0][j] = __builtin_amdgcn_wmma_f32_16x16x32_bf16(
          false, a0, false, b, (short)0, acc[0][j], false, false);
      acc[1][j] = __builtin_amdgcn_wmma_f32_16x16x32_bf16(
          false, a1, false, b, (short)0, acc[1][j], false, false);
    }
  }
  // epilogue: C layout -> lane holds column n (fixed), rows (strip*16)+lh*8+r
#pragma unroll
  for (int j = 0; j < 4; ++j) {
    const int n = n0 + j * 16 + lm;
    const float bv = gloadf(bias + n);
#pragma unroll
    for (int s = 0; s < 2; ++s) {
#pragma unroll
      for (int r = 0; r < 8; ++r) {
        const int m = s * 16 + lh * 8 + r;
        float c = acc[s][j][r] + bv;
        bf16_t* dp = D + m * sD + n;
        if (MODE == EP_RELU)      c = fmaxf(c, 0.0f);
        else if (MODE == EP_SIG)  c = 1.0f / (1.0f + __expf(-c));
        else if (MODE == EP_ADD)  c += (float)(*dp);
        *dp = (bf16_t)c;
      }
    }
  }
}

struct Params {
  const float *text, *num;
  const float *lnts, *lntb, *lnns, *lnnb;
  const float *b_tp, *b_np;
  const float *bv_ts, *bo_ts, *bv_ns, *bo_ns, *bv_tn, *bo_tn, *bv_nt, *bo_nt;
  const float *b_g, *b_f1, *b_f2, *b_o;
  const bf16_t *Wtp, *Wnp, *Wv_ts, *Wo_ts, *Wv_ns, *Wo_ns, *Wv_tn, *Wo_tn,
               *Wv_nt, *Wo_nt, *Wg, *Wf1, *Wf2, *Wo3;
  float* out;
};

__global__ __launch_bounds__(256, 1)
void HybridFusion_fused_kernel(Params p)
{
  // 6 x [32][520] bf16 activation buffers = 199,680 B of the 320 KB WGP LDS.
  __shared__ __attribute__((aligned(16))) bf16_t smem[6 * MT * S];
  bf16_t* tp  = smem;              // tp, later h2
  bf16_t* np_ = smem + 1 * MT * S; // np
  bf16_t* ts  = smem + 2 * MT * S; // t_self -> tc -> g*tc
  bf16_t* ns  = smem + 3 * MT * S; // n_self -> nc -> (1-g)*nc
  bf16_t* v   = smem + 4 * MT * S; // scratch (v_ts, v_tn, gate g)
  bf16_t* v2  = smem + 5 * MT * S; // scratch (v_ns, v_nt, h1)
  // LN staging aliases v/v2 (consumed before v is first written)
  bf16_t* tn_buf = v;                  // [32][776]
  bf16_t* nn_buf = v + MT * TSTR;      // [32][72], lives inside v2 region

  const int tid  = threadIdx.x;
  const int lane = tid & 31, wave = tid >> 5;
  const size_t row0 = (size_t)blockIdx.x * MT;

  // ---------- LayerNorm (text 768, num 64) -> bf16 staging ----------
  for (int i = 0; i < 4; ++i) {
    const int r = wave * 4 + i;
    const float* x = p.text + (row0 + r) * 768;
    float xs[24];
    float s = 0.f, sq = 0.f;
#pragma unroll
    for (int t = 0; t < 24; ++t) {
      xs[t] = gloadf(x + lane + 32 * t);
      s += xs[t]; sq += xs[t] * xs[t];
    }
    for (int off = 16; off; off >>= 1) { s += __shfl_xor(s, off); sq += __shfl_xor(sq, off); }
    const float mean = s * (1.f / 768.f);
    const float rstd = rsqrtf(sq * (1.f / 768.f) - mean * mean + 1e-5f);
#pragma unroll
    for (int t = 0; t < 24; ++t) {
      const int c = lane + 32 * t;
      tn_buf[r * TSTR + c] =
          (bf16_t)((xs[t] - mean) * rstd * gloadf(p.lnts + c) + gloadf(p.lntb + c));
    }

    const float* y = p.num + (row0 + r) * 64;
    float y0 = gloadf(y + lane), y1 = gloadf(y + lane + 32);
    float s2 = y0 + y1, q2 = y0 * y0 + y1 * y1;
    for (int off = 16; off; off >>= 1) { s2 += __shfl_xor(s2, off); q2 += __shfl_xor(q2, off); }
    const float mn = s2 * (1.f / 64.f);
    const float rs = rsqrtf(q2 * (1.f / 64.f) - mn * mn + 1e-5f);
    nn_buf[r * NSTR + lane] =
        (bf16_t)((y0 - mn) * rs * gloadf(p.lnns + lane) + gloadf(p.lnnb + lane));
    nn_buf[r * NSTR + lane + 32] =
        (bf16_t)((y1 - mn) * rs * gloadf(p.lnns + lane + 32) + gloadf(p.lnnb + lane + 32));
  }
  __syncthreads();

  // ---------- input projections ----------
  gemm_tile<768, 768, EP_NONE>(tn_buf, TSTR, tn_buf, TSTR, p.Wtp, p.b_tp, tp,  S, tid);
  gemm_tile< 64,  64, EP_NONE>(nn_buf, NSTR, nn_buf, NSTR, p.Wnp, p.b_np, np_, S, tid);
  __syncthreads();

  // ---------- self attention (len-1 softmax == 1  =>  ctx == v) ----------
  gemm_tile<512, 512, EP_NONE>(tp,  S, tp,  S, p.Wv_ts, p.bv_ts, v,  S, tid);
  gemm_tile<512, 512, EP_NONE>(np_, S, np_, S, p.Wv_ns, p.bv_ns, v2, S, tid);
  __syncthreads();
  gemm_tile<512, 512, EP_NONE>(v,  S, v,  S, p.Wo_ts, p.bo_ts, ts, S, tid);  // t_self
  gemm_tile<512, 512, EP_NONE>(v2, S, v2, S, p.Wo_ns, p.bo_ns, ns, S, tid);  // n_self
  __syncthreads();

  // ---------- cross attention ----------
  gemm_tile<512, 512, EP_NONE>(ns, S, ns, S, p.Wv_tn, p.bv_tn, v,  S, tid);  // v_tn(n_self)
  gemm_tile<512, 512, EP_NONE>(ts, S, ts, S, p.Wv_nt, p.bv_nt, v2, S, tid);  // v_nt(t_self)
  __syncthreads();
  gemm_tile<512, 512, EP_ADD>(v,  S, v,  S, p.Wo_tn, p.bo_tn, ts, S, tid);   // tc = t_self + t_x
  gemm_tile<512, 512, EP_ADD>(v2, S, v2, S, p.Wo_nt, p.bo_nt, ns, S, tid);   // nc = n_self + n_x
  __syncthreads();

  // ---------- gate: sigmoid([tc,nc] @ Wg^T + b) ----------
  gemm_tile<1024, 512, EP_SIG>(ts, S, ns, S, p.Wg, p.b_g, v, S, tid);
  __syncthreads();
  for (int idx = tid; idx < MT * 512; idx += 256) {
    const int m = idx >> 9, n = idx & 511;
    const float g = (float)v[m * S + n];
    const float a = (float)ts[m * S + n];
    const float b = (float)ns[m * S + n];
    ts[m * S + n] = (bf16_t)(g * a);
    ns[m * S + n] = (bf16_t)((1.f - g) * b);
  }
  __syncthreads();

  // ---------- FFN ----------
  gemm_tile<1024, 512, EP_RELU>(ts, S, ns, S, p.Wf1, p.b_f1, v2, S, tid);
  __syncthreads();
  gemm_tile<512, 512, EP_RELU>(v2, S, v2, S, p.Wf2, p.b_f2, tp, S, tid);
  __syncthreads();

  // ---------- head (N=3): scalar dot ----------
  if (tid < 96) {
    const int m = tid / 3, j = tid % 3;
    const bf16_t* h = tp + m * S;
    const bf16_t* w = p.Wo3 + j * 512;
    float acc = gloadf(p.b_o + j);
#pragma unroll 8
    for (int k = 0; k < 512; ++k) acc += (float)h[k] * gloadbf(w + k);
    gstoref(p.out + (row0 + m) * 3 + j, acc);
  }
}

__global__ void cvt_bf16_kernel(const float* __restrict__ src, bf16_t* __restrict__ dst, int n) {
  const int i = blockIdx.x * 256 + threadIdx.x;
  if (i < n) *(g_bf16)(uintptr_t)(dst + i) = (bf16_t)gloadf(src + i);
}

extern "C" void kernel_launch(void* const* d_in, const int* in_sizes, int n_in,
                              void* d_out, int out_size, void* d_ws, size_t ws_size,
                              hipStream_t stream)
{
  (void)in_sizes; (void)n_in; (void)out_size; (void)ws_size;
  auto F = [&](int i) { return (const float*)d_in[i]; };

  char* ws = (char*)d_ws;
  size_t off = 0;
  auto alloc_bf16 = [&](size_t nel) { bf16_t* p = (bf16_t*)(ws + off); off += nel * 2; return p; };

  bf16_t* Wtp   = alloc_bf16(512 * 768);
  bf16_t* Wnp   = alloc_bf16(512 * 64);
  bf16_t* Wv_ts = alloc_bf16(512 * 512);
  bf16_t* Wo_ts = alloc_bf16(512 * 512);
  bf16_t* Wv_ns = alloc_bf16(512 * 512);
  bf16_t* Wo_ns = alloc_bf16(512 * 512);
  bf16_t* Wv_tn = alloc_bf16(512 * 512);
  bf16_t* Wo_tn = alloc_bf16(512 * 512);
  bf16_t* Wv_nt = alloc_bf16(512 * 512);
  bf16_t* Wo_nt = alloc_bf16(512 * 512);
  bf16_t* Wg    = alloc_bf16(512 * 1024);
  bf16_t* Wf1   = alloc_bf16(512 * 1024);
  bf16_t* Wf2   = alloc_bf16(512 * 512);
  bf16_t* Wo3   = alloc_bf16(3 * 512);

  auto cvt = [&](const float* s, bf16_t* d, int n) {
    cvt_bf16_kernel<<<(n + 255) / 256, 256, 0, stream>>>(s, d, n);
  };
  const int VOFF = 2 * 512 * 512;  // v-slice of packed q/k/v in_proj
  cvt(F(6),  Wtp, 512 * 768);
  cvt(F(8),  Wnp, 512 * 64);
  cvt(F(10) + VOFF, Wv_ts, 512 * 512); cvt(F(12), Wo_ts, 512 * 512);
  cvt(F(14) + VOFF, Wv_ns, 512 * 512); cvt(F(16), Wo_ns, 512 * 512);
  cvt(F(18) + VOFF, Wv_tn, 512 * 512); cvt(F(20), Wo_tn, 512 * 512);
  cvt(F(22) + VOFF, Wv_nt, 512 * 512); cvt(F(24), Wo_nt, 512 * 512);
  cvt(F(26), Wg,  512 * 1024);
  cvt(F(28), Wf1, 512 * 1024);
  cvt(F(30), Wf2, 512 * 512);
  cvt(F(32), Wo3, 3 * 512);

  Params p{};
  p.text = F(0);  p.num  = F(1);
  p.lnts = F(2);  p.lntb = F(3);  p.lnns = F(4);  p.lnnb = F(5);
  p.b_tp = F(7);  p.b_np = F(9);
  p.bv_ts = F(11) + 1024; p.bo_ts = F(13);
  p.bv_ns = F(15) + 1024; p.bo_ns = F(17);
  p.bv_tn = F(19) + 1024; p.bo_tn = F(21);
  p.bv_nt = F(23) + 1024; p.bo_nt = F(25);
  p.b_g = F(27); p.b_f1 = F(29); p.b_f2 = F(31); p.b_o = F(33);
  p.Wtp = Wtp; p.Wnp = Wnp;
  p.Wv_ts = Wv_ts; p.Wo_ts = Wo_ts; p.Wv_ns = Wv_ns; p.Wo_ns = Wo_ns;
  p.Wv_tn = Wv_tn; p.Wo_tn = Wo_tn; p.Wv_nt = Wv_nt; p.Wo_nt = Wo_nt;
  p.Wg = Wg; p.Wf1 = Wf1; p.Wf2 = Wf2; p.Wo3 = Wo3;
  p.out = (float*)d_out;

  HybridFusion_fused_kernel<<<65536 / MT, 256, 0, stream>>>(p);
}
